// MSA_27169963114697
// MI455X (gfx1250) — matrix operand order repaired and checked by reading, not compiled, not run on hardware
//
#include <hip/hip_runtime.h>
#include <hip/hip_bf16.h>

// ---------------------------------------------------------------------------
// MI455X (gfx1250) windowed multi-head attention, bf16 WMMA pipeline with
// CDNA5 async global->LDS staging (global_load_async_to_lds_b128 + ASYNCcnt).
//   Stage 1: f32 -> bf16 conversions (x, qkv_w, proj_w); mask*edge fusion
//   Stage 2: QKV GEMM, 32x64 wave tiles, 64-K LDS panels double-buffered,
//            fully unrolled K loop with software-pipelined LDS fragments
//   Stage 3: fused scores + masked negated softmax + P@V (K/V^T/P in LDS)
//   Stage 4: output projection GEMM, same pipeline, f32 result + bias
// ---------------------------------------------------------------------------

typedef __bf16 bf16_t;
typedef __attribute__((ext_vector_type(16))) __bf16 v16bf;
typedef __attribute__((ext_vector_type(4)))  __bf16 v4bf;
typedef __attribute__((ext_vector_type(8)))  float  v8f;
typedef __attribute__((ext_vector_type(4)))  float  v4f;

#define DIM      768
#define HEADS    12
#define HEAD_DIM 64
#define NTOK     128
#define NWIN     3072            // B*f*H = 256*12
#define NTOKENS  32768           // B*f*NTOK
#define QKV_N    2304
#define SCALE_F  0.125f          // 64^-0.5

__device__ __forceinline__ v8f wmma_bf16(v16bf a, v16bf b, v8f c) {
  return __builtin_amdgcn_wmma_f32_16x16x32_bf16(
      /*neg_a=*/false, a, /*neg_b=*/false, b,
      /*c_mod=*/(short)0, c, /*reuse_a=*/false, /*reuse_b=*/false);
}

// --- CDNA5 async global->LDS copy (16B per lane), tracked by ASYNCcnt ------
__device__ __forceinline__ uint32_t lds_addr_of(const void* p) {
  return (uint32_t)(uintptr_t)p;   // low 32 bits of generic = LDS byte offset
}
__device__ __forceinline__ void async_ld_b128(uint32_t lds_off, const bf16_t* g) {
  asm volatile("global_load_async_to_lds_b128 %0, %1, off"
               :: "v"(lds_off), "v"(g) : "memory");
}
__device__ __forceinline__ void wait_async0() {
  asm volatile("s_wait_asynccnt 0x0" ::: "memory");
}

// A fragment (16x32 bf16, row-major, stride lda). Contiguous K: lane half h holds K 16h..16h+15.
__device__ __forceinline__ v16bf load_a_frag(const bf16_t* A, int lda, int lane) {
  const bf16_t* p = A + (size_t)(lane & 15) * lda + ((lane >> 4) << 4);
  v16bf a;
#pragma unroll
  for (int i = 0; i < 8; ++i) a[i] = p[i];
#pragma unroll
  for (int i = 0; i < 8; ++i) a[8 + i] = p[8 + i];
  return a;
}

// B fragment: 16 contiguous bf16 per lane (lanes 0-15: K 0..15 of column
// lane&15; lanes 16-31: K 16..31).
__device__ __forceinline__ v16bf load_bvec16(const bf16_t* p) {
  v16bf b;
#pragma unroll
  for (int i = 0; i < 16; ++i) b[i] = p[i];
  return b;
}

// ---------------------------------------------------------------------------
// Stage 1 kernels
// ---------------------------------------------------------------------------
__global__ void cvt_f32_bf16(const float* __restrict__ src,
                             bf16_t* __restrict__ dst, int n4) {
  int i = blockIdx.x * blockDim.x + threadIdx.x;
  if (i < n4) {
    v4f f = ((const v4f*)src)[i];
    v4bf b;
#pragma unroll
    for (int j = 0; j < 4; ++j) b[j] = (bf16_t)f[j];
    ((v4bf*)dst)[i] = b;
  }
}

__global__ void fuse_mask_edge(const float* __restrict__ mask,
                               const float* __restrict__ edge,
                               float* __restrict__ me, int n4) {
  int i = blockIdx.x * blockDim.x + threadIdx.x;
  if (i < n4) {
    v4f m = ((const v4f*)mask)[i];
    v4f e = ((const v4f*)edge)[i & 4095];  // 128*128/4 per frame
    ((v4f*)me)[i] = m * e;
  }
}

// ---------------------------------------------------------------------------
// GEMM core: block tile 256x64, wave tile 32x64 (8 accums, 2 A row-halves),
// 64-wide K panels double-buffered in LDS via async copies. Step loop fully
// unrolled (NSTEP compile-time) so A double-buffering is pure SSA renaming;
// B fragments software-pipelined one pair ahead.
// ---------------------------------------------------------------------------
#define B_LD 72   // padded LDS row stride (bf16 elems) for 64-wide K panels

template <int NSTEP>
__device__ __forceinline__ void gemm_core(
    const bf16_t* __restrict__ Abase, const bf16_t* __restrict__ gB,
    bf16_t (*Bs)[64 * B_LD], const uint32_t* ldsB, int lane, v8f acc[8]) {
  const int col  = lane & 15;
  const int khiv = (lane >> 4) << 4;

  async_ld_b128(ldsB[0], gB);
  async_ld_b128(ldsB[0] + 16, gB + 8);
  v16bf Ac[4];
  Ac[0] = load_a_frag(Abase, DIM, lane);
  Ac[1] = load_a_frag(Abase + 16 * DIM, DIM, lane);
  Ac[2] = load_a_frag(Abase + 32, DIM, lane);
  Ac[3] = load_a_frag(Abase + 16 * DIM + 32, DIM, lane);
  wait_async0();
  __syncthreads();

#pragma unroll
  for (int step = 0; step < NSTEP; ++step) {
    const int cur = step & 1;
    v16bf An[4];
    if (step + 1 < NSTEP) {
      const int kn = (step + 1) * 64;
      async_ld_b128(ldsB[cur ^ 1], gB + kn);
      async_ld_b128(ldsB[cur ^ 1] + 16, gB + kn + 8);
      An[0] = load_a_frag(Abase + kn, DIM, lane);
      An[1] = load_a_frag(Abase + 16 * DIM + kn, DIM, lane);
      An[2] = load_a_frag(Abase + kn + 32, DIM, lane);
      An[3] = load_a_frag(Abase + 16 * DIM + kn + 32, DIM, lane);
    }
    // four (khalf, pair) groups, pipelined one group ahead
    v16bf c0 = load_bvec16(&Bs[cur][(0 * 16 + col) * B_LD + khiv]);
    v16bf c1 = load_bvec16(&Bs[cur][(1 * 16 + col) * B_LD + khiv]);
#pragma unroll
    for (int pp = 0; pp < 4; ++pp) {
      const int kh = pp >> 1;   // K half: 0,0,1,1
      const int p  = pp & 1;    // column pair: 0,1,0,1
      v16bf d0 = c0, d1 = c1;
      if (pp < 3) {
        const int nkh = (pp + 1) >> 1, np = (pp + 1) & 1;
        d0 = load_bvec16(&Bs[cur][((np * 2) * 16 + col) * B_LD + nkh * 32 + khiv]);
        d1 = load_bvec16(&Bs[cur][((np * 2 + 1) * 16 + col) * B_LD + nkh * 32 + khiv]);
      }
      const int t0 = p * 2, t1 = p * 2 + 1;
      acc[t0]     = wmma_bf16(Ac[kh * 2],     c0, acc[t0]);
      acc[t0 + 4] = wmma_bf16(Ac[kh * 2 + 1], c0, acc[t0 + 4]);
      acc[t1]     = wmma_bf16(Ac[kh * 2],     c1, acc[t1]);
      acc[t1 + 4] = wmma_bf16(Ac[kh * 2 + 1], c1, acc[t1 + 4]);
      c0 = d0;
      c1 = d1;
    }
    if (step + 1 < NSTEP) {
      wait_async0();
      __syncthreads();
#pragma unroll
      for (int i = 0; i < 4; ++i) Ac[i] = An[i];  // SSA rename (loop unrolled)
    }
  }
}

// ---------------------------------------------------------------------------
// Stage 2: QKV GEMM. Grid (NTOKENS/256, QKV_N/64), 256 threads (8 waves).
// ---------------------------------------------------------------------------
__global__ __launch_bounds__(256) void qkv_gemm(
    const bf16_t* __restrict__ xb, const bf16_t* __restrict__ wb,
    const float* __restrict__ bias,
    bf16_t* __restrict__ qb, bf16_t* __restrict__ kb,
    bf16_t* __restrict__ vtb) {
  __shared__ bf16_t Bs[2][64 * B_LD];

  const int lane = threadIdx.x & 31;
  const int wave = threadIdx.x >> 5;
  const int m0 = blockIdx.x * 256 + wave * 32;
  const int n0 = blockIdx.y * 64;

  const int tn = threadIdx.x >> 2;        // staging: panel row 0..63
  const int tk = (threadIdx.x & 3) << 4;  // 0,16,32,48 (two b128 each)
  const bf16_t* gB = wb + (size_t)(n0 + tn) * DIM + tk;
  const uint32_t ldsB[2] = { lds_addr_of(&Bs[0][tn * B_LD + tk]),
                             lds_addr_of(&Bs[1][tn * B_LD + tk]) };

  v8f acc[8] = {};
  gemm_core<DIM / 64>(xb + (size_t)m0 * DIM, gB, Bs, ldsB, lane, acc);

  const int which = n0 / DIM;          // 0=q 1=k 2=v (uniform per block)
  const int h     = (n0 % DIM) >> 6;   // head (uniform per block)
#pragma unroll
  for (int t = 0; t < 4; ++t) {
    const int nc = n0 + t * 16 + (lane & 15);
    const float bv = bias[nc];
    const int d = nc & 63;
#pragma unroll
    for (int rh = 0; rh < 2; ++rh) {
#pragma unroll
      for (int r = 0; r < 8; ++r) {
        const int m = m0 + rh * 16 + ((lane >> 4) << 3) + r;  // global token
        const int win = (m >> 7) * HEADS + h;
        const int tok = m & 127;
        const float v = acc[rh * 4 + t][r] + bv;
        if (which == 0)      qb [((size_t)win * NTOK + tok) * HEAD_DIM + d] = (bf16_t)(v * SCALE_F);
        else if (which == 1) kb [((size_t)win * NTOK + tok) * HEAD_DIM + d] = (bf16_t)v;
        else                 vtb[((size_t)win * HEAD_DIM + d) * NTOK + tok] = (bf16_t)v; // V^T
      }
    }
  }
}

// ---------------------------------------------------------------------------
// Stage 3: fused attention per (b_f, head). K and V^T staged in LDS by async
// copies; P staged in LDS between score-softmax and P@V.
// ---------------------------------------------------------------------------
#define K_LD 72    // padded row stride for K tile  (64-wide rows)
#define V_LD 136   // padded row stride for V^T tile (128-wide rows)
#define P_LD 136   // padded row stride for P strips

__global__ __launch_bounds__(256) void attn_fused(
    const bf16_t* __restrict__ qb, const bf16_t* __restrict__ kb,
    const bf16_t* __restrict__ vtb, const float* __restrict__ me,
    bf16_t* __restrict__ aout) {
  __shared__ bf16_t Kst[NTOK * K_LD];       // 9216 elems
  __shared__ bf16_t Vst[HEAD_DIM * V_LD];   //  8704 elems
  __shared__ bf16_t Pl[8 * 16 * P_LD];      // 17408 elems

  const int lane = threadIdx.x & 31;
  const int wave = threadIdx.x >> 5;
  const int win  = blockIdx.x;
  const int bf   = win / HEADS;
  const int h    = win % HEADS;
  const int frame = bf & 63;               // b_f = b*FRAMES + f, FRAMES=64

  const bf16_t* q  = qb  + (size_t)win * NTOK * HEAD_DIM;
  const bf16_t* gK = kb  + (size_t)win * NTOK * HEAD_DIM;
  const bf16_t* gV = vtb + (size_t)win * HEAD_DIM * NTOK;
  const float* merow = me + (size_t)frame * NTOK * NTOK;

  // --- async stage K (128x64) and V^T (64x128) into padded LDS tiles ---
  for (int c = threadIdx.x; c < 1024; c += 256) {       // 1024 chunks of 16B
    const int kr = c >> 3, kc = (c & 7) << 3;           // K: 8 chunks/row
    async_ld_b128(lds_addr_of(&Kst[kr * K_LD + kc]), gK + kr * HEAD_DIM + kc);
    const int vr = c >> 4, vc = (c & 15) << 3;          // V: 16 chunks/row
    async_ld_b128(lds_addr_of(&Vst[vr * V_LD + vc]), gV + vr * NTOK + vc);
  }

  const int m0 = wave * 16;
  v16bf a0 = load_a_frag(q + (size_t)m0 * HEAD_DIM, HEAD_DIM, lane);
  v16bf a1 = load_a_frag(q + (size_t)m0 * HEAD_DIM + 32, HEAD_DIM, lane);

  wait_async0();
  __syncthreads();

  // --- scores: 16x128 strip, software-pipelined K fragments from LDS ---
  const int khi = (lane >> 4) << 4;
  v8f s[8] = {};
  v16bf b0 = load_bvec16(&Kst[(lane & 15) * K_LD + khi]);
  v16bf b1 = load_bvec16(&Kst[(lane & 15) * K_LD + 32 + khi]);
#pragma unroll
  for (int t = 0; t < 8; ++t) {
    v16bf nb0 = b0, nb1 = b1;
    if (t < 7) {
      const int row = (t + 1) * 16 + (lane & 15);
      nb0 = load_bvec16(&Kst[row * K_LD + khi]);
      nb1 = load_bvec16(&Kst[row * K_LD + 32 + khi]);
    }
    s[t] = wmma_bf16(a0, b0, s[t]);
    s[t] = wmma_bf16(a1, b1, s[t]);
    b0 = nb0;
    b1 = nb1;
  }

  // --- (mask*edge), masked_fill(==0,-1e4), z=-score, row max/sum ---
  float rmax[8];
#pragma unroll
  for (int r = 0; r < 8; ++r) rmax[r] = -3.4e38f;
#pragma unroll
  for (int t = 0; t < 8; ++t) {
    const int n = t * 16 + (lane & 15);
#pragma unroll
    for (int r = 0; r < 8; ++r) {
      const int m = m0 + ((lane >> 4) << 3) + r;
      const float sv = s[t][r] * merow[m * NTOK + n];
      const float z = (sv == 0.0f) ? 10000.0f : -sv;  // -(masked_fill)
      s[t][r] = z;
      rmax[r] = fmaxf(rmax[r], z);
    }
  }
#pragma unroll
  for (int r = 0; r < 8; ++r) {
#pragma unroll
    for (int off = 1; off < 16; off <<= 1)
      rmax[r] = fmaxf(rmax[r], __shfl_xor(rmax[r], off, 32));
  }
  float rsum[8];
#pragma unroll
  for (int r = 0; r < 8; ++r) rsum[r] = 0.0f;
#pragma unroll
  for (int t = 0; t < 8; ++t) {
#pragma unroll
    for (int r = 0; r < 8; ++r) {
      const float e = __expf(s[t][r] - rmax[r]);
      s[t][r] = e;
      rsum[r] += e;
    }
  }
#pragma unroll
  for (int r = 0; r < 8; ++r) {
#pragma unroll
    for (int off = 1; off < 16; off <<= 1)
      rsum[r] += __shfl_xor(rsum[r], off, 32);
    rsum[r] = 1.0f / rsum[r];
  }

  // --- normalized P (bf16) -> LDS (per-wave strip, A-fragment layout) ---
  bf16_t* Pw = Pl + wave * 16 * P_LD;
#pragma unroll
  for (int t = 0; t < 8; ++t) {
    const int n = t * 16 + (lane & 15);
#pragma unroll
    for (int r = 0; r < 8; ++r) {
      const int m = ((lane >> 4) << 3) + r;
      Pw[m * P_LD + n] = (bf16_t)(s[t][r] * rsum[r]);
    }
  }
  // Each wave reads back only its own strip; per-wave DS ordering suffices.

  // --- P @ V : 16x64 output, K=128 in four steps, A and B from LDS ---
  v8f o[4] = {};
#pragma unroll
  for (int k0 = 0; k0 < NTOK; k0 += 32) {
    v16bf a = load_a_frag(Pw + k0, P_LD, lane);
    v16bf vb0 = load_bvec16(&Vst[(0 * 16 + (lane & 15)) * V_LD + k0 + khi]);
    v16bf vb1 = load_bvec16(&Vst[(1 * 16 + (lane & 15)) * V_LD + k0 + khi]);
    v16bf vb2 = load_bvec16(&Vst[(2 * 16 + (lane & 15)) * V_LD + k0 + khi]);
    v16bf vb3 = load_bvec16(&Vst[(3 * 16 + (lane & 15)) * V_LD + k0 + khi]);
    o[0] = wmma_bf16(a, vb0, o[0]);
    o[1] = wmma_bf16(a, vb1, o[1]);
    o[2] = wmma_bf16(a, vb2, o[2]);
    o[3] = wmma_bf16(a, vb3, o[3]);
  }

  // --- store attention output as [token][DIM] bf16 for the proj GEMM ---
#pragma unroll
  for (int t = 0; t < 4; ++t) {
    const int d = t * 16 + (lane & 15);
#pragma unroll
    for (int r = 0; r < 8; ++r) {
      const int m = m0 + ((lane >> 4) << 3) + r;
      const int token = bf * NTOK + m;
      aout[(size_t)token * DIM + h * HEAD_DIM + d] = (bf16_t)o[t][r];
    }
  }
}

// ---------------------------------------------------------------------------
// Stage 4: output projection, f32 result. Grid (NTOKENS/256, DIM/64).
// ---------------------------------------------------------------------------
__global__ __launch_bounds__(256) void proj_gemm(
    const bf16_t* __restrict__ ab, const bf16_t* __restrict__ wb,
    const float* __restrict__ bias, float* __restrict__ out) {
  __shared__ bf16_t Bs[2][64 * B_LD];

  const int lane = threadIdx.x & 31;
  const int wave = threadIdx.x >> 5;
  const int m0 = blockIdx.x * 256 + wave * 32;
  const int n0 = blockIdx.y * 64;

  const int tn = threadIdx.x >> 2;
  const int tk = (threadIdx.x & 3) << 4;
  const bf16_t* gB = wb + (size_t)(n0 + tn) * DIM + tk;
  const uint32_t ldsB[2] = { lds_addr_of(&Bs[0][tn * B_LD + tk]),
                             lds_addr_of(&Bs[1][tn * B_LD + tk]) };

  v8f acc[8] = {};
  gemm_core<DIM / 64>(ab + (size_t)m0 * DIM, gB, Bs, ldsB, lane, acc);

#pragma unroll
  for (int t = 0; t < 4; ++t) {
    const int n = n0 + t * 16 + (lane & 15);
    const float bv = bias[n];
#pragma unroll
    for (int rh = 0; rh < 2; ++rh) {
#pragma unroll
      for (int r = 0; r < 8; ++r) {
        const int m = m0 + rh * 16 + ((lane >> 4) << 3) + r;
        out[(size_t)m * DIM + n] = acc[rh * 4 + t][r] + bv;
      }
    }
  }
}

// ---------------------------------------------------------------------------
extern "C" void kernel_launch(void* const* d_in, const int* in_sizes, int n_in,
                              void* d_out, int out_size, void* d_ws, size_t ws_size,
                              hipStream_t stream) {
  (void)in_sizes; (void)n_in; (void)out_size; (void)ws_size;
  const float* x      = (const float*)d_in[0];   // [B_f*N, DIM] flat
  const float* mask   = (const float*)d_in[1];   // [64,128,128]
  const float* edge   = (const float*)d_in[2];   // [128,128]
  const float* qkv_w  = (const float*)d_in[3];   // [2304,768]
  const float* qkv_b  = (const float*)d_in[4];   // [2304]
  const float* proj_w = (const float*)d_in[5];   // [768,768]
  const float* proj_b = (const float*)d_in[6];   // [768]
  float* out = (float*)d_out;                    // [32768,768]

  char* ws = (char*)d_ws;
  size_t off = 0;
  auto carve = [&](size_t bytes) {
    char* p = ws + off;
    off += (bytes + 255) & ~(size_t)255;
    return p;
  };
  bf16_t* xb  = (bf16_t*)carve((size_t)NTOKENS * DIM * 2);          // 50.3 MB
  bf16_t* qwb = (bf16_t*)carve((size_t)QKV_N * DIM * 2);            //  3.5 MB
  bf16_t* pwb = (bf16_t*)carve((size_t)DIM * DIM * 2);              //  1.2 MB
  bf16_t* qb  = (bf16_t*)carve((size_t)NWIN * NTOK * HEAD_DIM * 2); // 50.3 MB
  bf16_t* kb  = (bf16_t*)carve((size_t)NWIN * NTOK * HEAD_DIM * 2); // 50.3 MB
  bf16_t* vtb = (bf16_t*)carve((size_t)NWIN * HEAD_DIM * NTOK * 2); // 50.3 MB
  float*  me  = (float*) carve((size_t)64 * NTOK * NTOK * 4);       //  4.2 MB
  bf16_t* aout = xb;  // alias: x(bf16) fully consumed by qkv_gemm before attn writes

  int n4;
  n4 = NTOKENS * DIM / 4;
  cvt_f32_bf16<<<(n4 + 255) / 256, 256, 0, stream>>>(x, xb, n4);
  n4 = QKV_N * DIM / 4;
  cvt_f32_bf16<<<(n4 + 255) / 256, 256, 0, stream>>>(qkv_w, qwb, n4);
  n4 = DIM * DIM / 4;
  cvt_f32_bf16<<<(n4 + 255) / 256, 256, 0, stream>>>(proj_w, pwb, n4);
  n4 = 64 * NTOK * NTOK / 4;
  fuse_mask_edge<<<(n4 + 255) / 256, 256, 0, stream>>>(mask, edge, me, n4);

  qkv_gemm<<<dim3(NTOKENS / 256, QKV_N / 64), 256, 0, stream>>>(
      xb, qwb, qkv_b, qb, kb, vtb);

  attn_fused<<<NWIN, 256, 0, stream>>>(qb, kb, vtb, me, aout);

  proj_gemm<<<dim3(NTOKENS / 256, DIM / 64), 256, 0, stream>>>(
      aout, pwb, proj_b, out);
}
